// HardMemory_39204461478033
// MI455X (gfx1250) — matrix operand order repaired and checked by reading, not compiled, not run on hardware
//
#include <hip/hip_runtime.h>
#include <hip/hip_bf16.h>

typedef __attribute__((ext_vector_type(16))) _Float16 v16h;
typedef __attribute__((ext_vector_type(8)))  float    v8f;

#define C_DIM     128
#define M_ROWS    512
#define HW        4096
#define B_DIM     32
#define NTOK      (B_DIM * HW)          // 131072 tokens
#define NTILES    (NTOK / 16)           // 8192 16-token tiles
#define NPAIRS    (NTILES / 2)          // 4096 32-token pair-tiles
#define MTILES    (M_ROWS / 16)         // 32 memory tiles
#define KCHUNKS   (C_DIM / 32)          // 4 K-chunks of 32

// ---------------------------------------------------------------------------
// Kernel 1: reciprocal L2 norm of each memory row (torch F.normalize eps)
// ---------------------------------------------------------------------------
__global__ __launch_bounds__(32) void mem_norms(const float* __restrict__ memory,
                                                float* __restrict__ rnorm) {
    const int row  = blockIdx.x;
    const int lane = threadIdx.x;
    float s = 0.f;
    #pragma unroll
    for (int k = 0; k < 4; ++k) {
        float v = memory[row * C_DIM + k * 32 + lane];
        s += v * v;
    }
    #pragma unroll
    for (int off = 16; off >= 1; off >>= 1)
        s += __shfl_xor(s, off, 32);
    if (lane == 0)
        rnorm[row] = 1.0f / fmaxf(sqrtf(s), 1e-12f);
}

// ---------------------------------------------------------------------------
// Kernel 2: pre-swizzle normalized memory into per-lane WMMA B-fragment layout
//   fragB[(mtile*KCHUNKS + chunk)*32 + lane][e] =
//       B[K = chunk*32 + e + 16*(lane/16)][N = mtile*16 + lane%16]
// ---------------------------------------------------------------------------
__global__ __launch_bounds__(32) void build_fragB(const float* __restrict__ memory,
                                                  const float* __restrict__ rnorm,
                                                  v16h* __restrict__ fragB) {
    const int mtile = blockIdx.x / KCHUNKS;
    const int chunk = blockIdx.x % KCHUNKS;
    const int lane  = threadIdx.x;
    const int l16   = lane & 15;
    const int hi    = lane >> 4;
    const int row   = mtile * 16 + l16;
    const float rn  = rnorm[row];
    v16h frag;
    #pragma unroll
    for (int e = 0; e < 16; ++e) {
        int ch = chunk * 32 + e + 16 * hi;
        frag[e] = (_Float16)(memory[row * C_DIM + ch] * rn);
    }
    fragB[blockIdx.x * 32 + lane] = frag;
}

// ---------------------------------------------------------------------------
// Kernel 3: 32 tokens per wave (two 16x16 WMMA row-tiles), running argmax over
// 512 memory rows. B fragments double-buffered in registers so L2 latency
// overlaps the WMMA pipe; two independent accumulator chains per fragment.
// ---------------------------------------------------------------------------
#define WMMA_F16(A_, B_, C_) \
    __builtin_amdgcn_wmma_f32_16x16x32_f16(false, (A_), false, (B_), (short)0, (C_), false, false)

#define COMPUTE_MT(BUF, MT_)                                                   \
    {                                                                          \
        v8f acc0 = {}; v8f acc1 = {};                                          \
        _Pragma("unroll")                                                      \
        for (int c = 0; c < KCHUNKS; ++c) {                                    \
            acc0 = WMMA_F16(a0[c], BUF[c], acc0);                              \
            acc1 = WMMA_F16(a1[c], BUF[c], acc1);                              \
        }                                                                      \
        const int col = (MT_) * 16 + l16;                                      \
        _Pragma("unroll")                                                      \
        for (int r = 0; r < 8; ++r) {                                          \
            if (acc0[r] > best0[r]) { best0[r] = acc0[r]; bidx0[r] = col; }    \
            if (acc1[r] > best1[r]) { best1[r] = acc1[r]; bidx1[r] = col; }    \
        }                                                                      \
    }

__global__ __launch_bounds__(128) void sim_argmax(const float* __restrict__ x,
                                                  const v16h* __restrict__ fragB,
                                                  int* __restrict__ idxOut) {
    const int lane  = threadIdx.x & 31;
    const int l16   = lane & 15;
    const int hi    = lane >> 4;
    const int wave  = (blockIdx.x * blockDim.x + threadIdx.x) >> 5;
    const int nwave = (gridDim.x * blockDim.x) >> 5;

    for (int pt = wave; pt < NPAIRS; pt += nwave) {
        const int t0  = pt << 5;                 // first global token (32/pair)
        const int b   = t0 >> 12;                // / HW (no straddle: 4096%32==0)
        const int hw0 = t0 & (HW - 1);
        const float* xb = x + (size_t)b * C_DIM * HW;

        // --- A fragments for both 16-token tiles (contiguous k = 16*hi + e, same as fragB) ---
        v16h a0[KCHUNKS], a1[KCHUNKS];
        #pragma unroll
        for (int c = 0; c < KCHUNKS; ++c) {
            #pragma unroll
            for (int e = 0; e < 16; ++e) {
                int ch = c * 32 + e + 16 * hi;
                const float* xc = xb + (size_t)ch * HW + hw0 + l16;
                a0[c][e] = (_Float16)xc[0];
                a1[c][e] = (_Float16)xc[16];
            }
        }

        float best0[8], best1[8];
        int   bidx0[8], bidx1[8];
        #pragma unroll
        for (int r = 0; r < 8; ++r) {
            best0[r] = -3.402823466e38f; bidx0[r] = 0;
            best1[r] = -3.402823466e38f; bidx1[r] = 0;
        }

        // --- double-buffered B fragments, 2 mtiles per iteration ---
        v16h bA[KCHUNKS], bB[KCHUNKS];
        #pragma unroll
        for (int c = 0; c < KCHUNKS; ++c)
            bA[c] = fragB[c * 32 + lane];                     // mt = 0

        for (int mt = 0; mt < MTILES; mt += 2) {
            #pragma unroll
            for (int c = 0; c < KCHUNKS; ++c)                 // prefetch mt+1
                bB[c] = fragB[((mt + 1) * KCHUNKS + c) * 32 + lane];

            COMPUTE_MT(bA, mt)

            if (mt + 2 < MTILES) {
                #pragma unroll
                for (int c = 0; c < KCHUNKS; ++c)             // prefetch mt+2
                    bA[c] = fragB[((mt + 2) * KCHUNKS + c) * 32 + lane];
            }

            COMPUTE_MT(bB, mt + 1)
        }

        // --- butterfly argmax across the 16 lanes of each half-wave ---
        #pragma unroll
        for (int off = 8; off >= 1; off >>= 1) {
            #pragma unroll
            for (int r = 0; r < 8; ++r) {
                float ov0 = __shfl_xor(best0[r], off, 32);
                int   oi0 = __shfl_xor(bidx0[r], off, 32);
                if (ov0 > best0[r] || (ov0 == best0[r] && oi0 < bidx0[r])) {
                    best0[r] = ov0; bidx0[r] = oi0;
                }
                float ov1 = __shfl_xor(best1[r], off, 32);
                int   oi1 = __shfl_xor(bidx1[r], off, 32);
                if (ov1 > best1[r] || (ov1 == best1[r] && oi1 < bidx1[r])) {
                    best1[r] = ov1; bidx1[r] = oi1;
                }
            }
        }

        if (l16 == 0) {                           // lanes 0 and 16 write 8 each
            #pragma unroll
            for (int r = 0; r < 8; ++r) {
                idxOut[t0 + 8 * hi + r]      = bidx0[r];
                idxOut[t0 + 16 + 8 * hi + r] = bidx1[r];
            }
        }
    }
}

// ---------------------------------------------------------------------------
// Kernel 4: out[b][c][n] = memory[idx[b][n]][c] ; flat out index == thread idx
// ---------------------------------------------------------------------------
__global__ __launch_bounds__(256) void gather_out(const float* __restrict__ memory,
                                                  const int* __restrict__ idx,
                                                  float* __restrict__ out) {
    const int i = blockIdx.x * 256 + threadIdx.x;
    const int n = i & (HW - 1);
    const int c = (i >> 12) & (C_DIM - 1);
    const int b = i >> 19;
    const int m = idx[b * HW + n];
    out[i] = memory[m * C_DIM + c];
}

// ---------------------------------------------------------------------------
extern "C" void kernel_launch(void* const* d_in, const int* in_sizes, int n_in,
                              void* d_out, int out_size, void* d_ws, size_t ws_size,
                              hipStream_t stream) {
    const float* x      = (const float*)d_in[0];   // (32,128,64,64) f32
    const float* memory = (const float*)d_in[1];   // (512,128) f32
    float* out          = (float*)d_out;           // (32,128,64,64) f32

    char* ws      = (char*)d_ws;
    float* rnorm  = (float*)ws;                          // 512 f32      (2 KiB)
    v16h*  fragB  = (v16h*)(ws + 4096);                  // 4096 v16h  (128 KiB)
    int*   idxBuf = (int*)(ws + 4096 + 131072);          // 131072 i32 (512 KiB)

    mem_norms  <<<M_ROWS, 32, 0, stream>>>(memory, rnorm);
    build_fragB<<<MTILES * KCHUNKS, 32, 0, stream>>>(memory, rnorm, fragB);
    sim_argmax <<<1024, 128, 0, stream>>>(x, fragB, idxBuf);  // 4096 waves, 1 pair each
    gather_out <<<(B_DIM * C_DIM * HW) / 256, 256, 0, stream>>>(memory, idxBuf, out);
}
